// MixedSparseSingleLayer_88347477279627
// MI455X (gfx1250) — matrix operand order repaired and checked by reading, not compiled, not run on hardware
//
#include <hip/hip_runtime.h>
#include <hip/hip_bf16.h>

typedef __attribute__((ext_vector_type(16))) _Float16 v16h;
typedef __attribute__((ext_vector_type(8)))  _Float16 v8h;
typedef __attribute__((ext_vector_type(8)))  float    v8f;

union Frag { v16h v; v8h h[2]; };

// ---------------------------------------------------------------------------
// Global -> LDS tile staging. Prefer the CDNA5 async direct-to-LDS path
// (GLOBAL_LOAD_ASYNC_TO_LDS_B128, tracked by ASYNCcnt) when the toolchain
// exposes the builtin; otherwise fall back to VGPR-staged b128 copies.
// The builtin signature (from the compiler diagnostic) is:
//   void __builtin_amdgcn_global_load_async_to_lds_b128(
//       int4 __device__* src, int4 __shared__* dst, imm int, imm int);
// ---------------------------------------------------------------------------
#if defined(__HIP_DEVICE_COMPILE__) && \
    __has_builtin(__builtin_amdgcn_global_load_async_to_lds_b128) && \
    __has_builtin(__builtin_amdgcn_s_wait_asynccnt)
#define USE_ASYNC_LDS 1
#else
#define USE_ASYNC_LDS 0
#endif

#if USE_ASYNC_LDS
typedef int v4i __attribute__((vector_size(16)));
typedef __attribute__((address_space(1))) v4i as1_v4i;
typedef __attribute__((address_space(3))) v4i as3_v4i;
#endif

__device__ __forceinline__ void copy16B(const _Float16* __restrict__ src,
                                        _Float16* __restrict__ dst) {
#if USE_ASYNC_LDS
  __builtin_amdgcn_global_load_async_to_lds_b128(
      (as1_v4i*)src, (as3_v4i*)dst, 0, 0);
#else
  *(uint4*)dst = *(const uint4*)src;
#endif
}

__device__ __forceinline__ void copy64B(const _Float16* __restrict__ src,
                                        _Float16* __restrict__ dst) {
#if USE_ASYNC_LDS
  #pragma unroll
  for (int i = 0; i < 4; ++i)
    __builtin_amdgcn_global_load_async_to_lds_b128(
        (as1_v4i*)(src + i * 8), (as3_v4i*)(dst + i * 8), 0, 0);
#else
  const uint4* s = (const uint4*)src;
  uint4* d = (uint4*)dst;
  d[0] = s[0]; d[1] = s[1]; d[2] = s[2]; d[3] = s[3];
#endif
}

__device__ __forceinline__ void async_join() {
#if USE_ASYNC_LDS
  __builtin_amdgcn_s_wait_asynccnt(0);
#endif
}

// ---------------------------------------------------------------------------
// Weight prep: Weff[n*K+k] = W[n*K+k] + sum_r la[k*R+r] * lb[r*N+n]  (f16 out)
// Folds LoRA into the dense weight so the GEMMs are pure WMMA.
// ---------------------------------------------------------------------------
__global__ __launch_bounds__(256) void prep_weight_kernel(
    const float* __restrict__ w, const float* __restrict__ la,
    const float* __restrict__ lb, _Float16* __restrict__ out,
    int Nn, int Kk, int Rr)
{
  const size_t idx = (size_t)blockIdx.x * 256 + threadIdx.x;
  const size_t total = (size_t)Nn * Kk;
  if (idx >= total) return;
  const int n = (int)(idx / Kk);
  const int k = (int)(idx % Kk);
  float acc = w[idx];
  for (int r = 0; r < Rr; ++r)
    acc += la[(size_t)k * Rr + r] * lb[(size_t)r * Nn + n];
  out[idx] = (_Float16)acc;
}

// ---------------------------------------------------------------------------
// LayerNorm (fp32 in -> f16 out), one block per row.
// ---------------------------------------------------------------------------
__global__ __launch_bounds__(256) void layernorm_kernel(
    const float* __restrict__ x, const float* __restrict__ gw,
    const float* __restrict__ gb, _Float16* __restrict__ out, int Hdim)
{
  __shared__ float r1[8], r2[8];
  const int row = blockIdx.x;
  const int tid = threadIdx.x;
  const float* xr = x + (size_t)row * Hdim;
  float s = 0.f, s2 = 0.f;
  for (int i = tid; i < Hdim; i += 256) { const float v = xr[i]; s += v; s2 += v * v; }
  #pragma unroll
  for (int m = 16; m >= 1; m >>= 1) { s += __shfl_xor(s, m, 32); s2 += __shfl_xor(s2, m, 32); }
  if ((tid & 31) == 0) { r1[tid >> 5] = s; r2[tid >> 5] = s2; }
  __syncthreads();
  float ts = 0.f, ts2 = 0.f;
  #pragma unroll
  for (int i = 0; i < 8; ++i) { ts += r1[i]; ts2 += r2[i]; }
  const float mean = ts / (float)Hdim;
  const float var  = ts2 / (float)Hdim - mean * mean;
  const float rinv = rsqrtf(var + 1e-10f);
  for (int i = tid; i < Hdim; i += 256)
    out[(size_t)row * Hdim + i] = (_Float16)((xr[i] - mean) * rinv * gw[i] + gb[i]);
}

// ---------------------------------------------------------------------------
// RoPE: per-head rotate-half on f16 activations [M, H], HD = 128.
// ---------------------------------------------------------------------------
__global__ __launch_bounds__(256) void rope_kernel(
    const _Float16* __restrict__ in, _Float16* __restrict__ out,
    const float* __restrict__ cs, const float* __restrict__ sn,
    int S, int Hdim, size_t total)
{
  const size_t idx = (size_t)blockIdx.x * 256 + threadIdx.x;
  if (idx >= total) return;
  const int m = (int)(idx / Hdim);
  const int c = (int)(idx % Hdim);
  const int s = m % S;
  const int d = c & 127;
  const float v = (float)in[idx];
  const float rot = (d < 64) ? -(float)in[idx + 64] : (float)in[idx - 64];
  out[idx] = (_Float16)(v * cs[(size_t)s * 128 + d] + rot * sn[(size_t)s * 128 + d]);
}

// ---------------------------------------------------------------------------
// WMMA GEMM: C[M,N] = A[M,K] (f16, row-major) x B[N,K] (f16, row-major, i.e.
// contraction over K per row) + bias, with fused epilogue.
// EPI 0: f16 out = acc + bias        (QKV projections)
// EPI 1: f16 out = silu(acc + bias)  (up projection)
// EPI 2: f32 out = acc + bias + res  (o / down projections with residual)
// Block: 256 threads = 8 waves, tile 128x128x64, double-buffered LDS,
// async global->LDS staging when available.
// ---------------------------------------------------------------------------
template<int EPI>
__global__ __launch_bounds__(256) void gemm_wmma(
    const _Float16* __restrict__ A, const _Float16* __restrict__ Bw,
    const float* __restrict__ bias, const float* __restrict__ res,
    void* __restrict__ outp, int Mm, int Nn, int Kk)
{
  constexpr int BM = 128, BN = 128, BK = 64;
  __shared__ __align__(16) _Float16 As[2][BM][BK];
  __shared__ __align__(16) _Float16 Bs[2][BN][BK];

  const int tid  = threadIdx.x;
  const int bm   = blockIdx.y * BM;
  const int bn   = blockIdx.x * BN;
  const int wid  = tid >> 5, lane = tid & 31;
  const int lh   = lane & 15, hi = lane >> 4;
  const int wm   = (wid & 3) * 32;   // wave M offset inside tile (2 x 16)
  const int wn   = (wid >> 2) * 64;  // wave N offset inside tile (4 x 16)
  const int lr   = tid >> 1;         // loader row (0..127)
  const int lc   = (tid & 1) * 32;   // loader col (halves)

  v8f acc[2][4];
  #pragma unroll
  for (int i = 0; i < 2; ++i)
    #pragma unroll
    for (int j = 0; j < 4; ++j)
      #pragma unroll
      for (int e = 0; e < 8; ++e) acc[i][j][e] = 0.0f;

  const int nkt = Kk / BK;

  copy64B(A  + (size_t)(bm + lr) * Kk + lc, &As[0][lr][lc]);
  copy64B(Bw + (size_t)(bn + lr) * Kk + lc, &Bs[0][lr][lc]);
  async_join();
  __syncthreads();

  for (int kt = 0; kt < nkt; ++kt) {
    const int cur = kt & 1;
    if (kt + 1 < nkt) {
      const size_t ko = (size_t)(kt + 1) * BK + lc;
      copy64B(A  + (size_t)(bm + lr) * Kk + ko, &As[cur ^ 1][lr][lc]);
      copy64B(Bw + (size_t)(bn + lr) * Kk + ko, &Bs[cur ^ 1][lr][lc]);
      if (kt + 2 < nkt) {
        __builtin_prefetch(A  + (size_t)(bm + lr) * Kk + ko + BK, 0, 1);
        __builtin_prefetch(Bw + (size_t)(bn + lr) * Kk + ko + BK, 0, 1);
      }
    }
    #pragma unroll
    for (int ks = 0; ks < 2; ++ks) {
      const int kb = ks * 32;
      Frag af[2], bf[4];
      #pragma unroll
      for (int i = 0; i < 2; ++i) {
        const _Float16* p = &As[cur][wm + i * 16 + lh][kb];
        af[i].h[0] = *(const v8h*)(p + hi * 8);        // K 0-7  / 8-15
        af[i].h[1] = *(const v8h*)(p + 16 + hi * 8);   // K 16-23 / 24-31
      }
      #pragma unroll
      for (int j = 0; j < 4; ++j) {
        const _Float16* p = &Bs[cur][wn + j * 16 + lh][kb];
        bf[j].h[0] = *(const v8h*)(p + hi * 8);        // K 0-7  / 8-15
        bf[j].h[1] = *(const v8h*)(p + 16 + hi * 8);   // K 16-23 / 24-31
      }
      #pragma unroll
      for (int i = 0; i < 2; ++i)
        #pragma unroll
        for (int j = 0; j < 4; ++j)
          acc[i][j] = __builtin_amdgcn_wmma_f32_16x16x32_f16(
              false, af[i].v, false, bf[j].v, (short)0, acc[i][j], false, false);
    }
    async_join();
    __syncthreads();
  }

  // Epilogue. C layout: VGPR e, lane l -> M = e + 8*(l>>4), N = l&15.
  #pragma unroll
  for (int i = 0; i < 2; ++i) {
    #pragma unroll
    for (int j = 0; j < 4; ++j) {
      const int col  = bn + wn + j * 16 + lh;
      const float bv = bias[col];
      const int row0 = bm + wm + i * 16 + hi * 8;
      #pragma unroll
      for (int e = 0; e < 8; ++e) {
        const size_t ridx = (size_t)(row0 + e) * Nn + col;
        float val = acc[i][j][e] + bv;
        if constexpr (EPI == 0) {
          ((_Float16*)outp)[ridx] = (_Float16)val;
        } else if constexpr (EPI == 1) {
          const float sg = 1.0f / (1.0f + __expf(-val));
          ((_Float16*)outp)[ridx] = (_Float16)(val * sg);
        } else {
          ((float*)outp)[ridx] = val + res[ridx];
        }
      }
    }
  }
}

// ---------------------------------------------------------------------------
// Flash attention (causal), HD=128. Block = 128 threads (4 waves), each wave
// owns 16 query rows; Br=64 queries per block, Bc=32 keys per step.
// Q*K^T and P*V both use v_wmma_f32_16x16x32_f16; online softmax per row.
// All B-fragments are preloaded into register arrays so the 8 WMMAs of each
// phase issue back-to-back instead of serializing on ds_load latency.
// ---------------------------------------------------------------------------
__global__ __launch_bounds__(128) void flash_attn_kernel(
    const _Float16* __restrict__ Q, const _Float16* __restrict__ Kp,
    const _Float16* __restrict__ V, _Float16* __restrict__ O,
    int S, int Hdim)
{
  constexpr int HD = 128, BR = 64, BC = 32;
  const int qt   = blockIdx.x;
  const int head = blockIdx.y;
  const int bb   = blockIdx.z;
  const int tid  = threadIdx.x;
  const int wid  = tid >> 5, lane = tid & 31;
  const int lh   = lane & 15, hi = lane >> 4;

  __shared__ __align__(16) _Float16 Ks[BC][HD];
  __shared__ __align__(16) _Float16 Vt[HD][BC];     // transposed V tile
  __shared__ __align__(16) _Float16 Ps[4][16][32];  // per-wave P staging

  const size_t base = ((size_t)bb * S) * Hdim + (size_t)head * HD;

  // Q fragments held in registers for the whole key loop (4 k-steps of 32).
  Frag qf[4];
  {
    const int qrow = qt * BR + wid * 16 + lh;
    const _Float16* qp = Q + base + (size_t)qrow * Hdim;
    #pragma unroll
    for (int kk = 0; kk < 4; ++kk) {
      qf[kk].h[0] = *(const v8h*)(qp + kk * 32 + hi * 8);
      qf[kk].h[1] = *(const v8h*)(qp + kk * 32 + 16 + hi * 8);
    }
  }

  v8f o_acc[8];
  #pragma unroll
  for (int nd = 0; nd < 8; ++nd)
    #pragma unroll
    for (int e = 0; e < 8; ++e) o_acc[nd][e] = 0.0f;
  float m_i[8], l_i[8];
  #pragma unroll
  for (int e = 0; e < 8; ++e) { m_i[e] = -3.0e38f; l_i[e] = 0.0f; }

  const float scale = 0.08838834764831845f; // 1/sqrt(128)
  const int nkt = 2 * (qt + 1);             // causal bound: keys <= (qt+1)*64

  for (int kt = 0; kt < nkt; ++kt) {
    const int k0 = kt * BC;
    __syncthreads();
    // K tile: [32 keys][128 d]  (async direct-to-LDS when available)
    for (int idx = tid; idx < BC * HD / 8; idx += 128) {
      const int r = idx >> 4;
      const int c = (idx & 15) * 8;
      copy16B(Kp + base + (size_t)(k0 + r) * Hdim + c, &Ks[r][c]);
    }
    // V tile transposed: Vt[d][key]
    for (int idx = tid; idx < BC * HD / 8; idx += 128) {
      const int r = idx >> 4;
      const int c = (idx & 15) * 8;
      const v8h tv = *(const v8h*)(V + base + (size_t)(k0 + r) * Hdim + c);
      #pragma unroll
      for (int e = 0; e < 8; ++e) Vt[c + e][r] = tv[e];
    }
    async_join();
    __syncthreads();

    // Preload all 8 K^T fragments, then issue 8 score WMMAs back-to-back.
    Frag kb[8];
    #pragma unroll
    for (int nt = 0; nt < 2; ++nt) {
      #pragma unroll
      for (int kk = 0; kk < 4; ++kk) {
        kb[nt * 4 + kk].h[0] = *(const v8h*)(&Ks[nt * 16 + lh][kk * 32 + hi * 8]);
        kb[nt * 4 + kk].h[1] = *(const v8h*)(&Ks[nt * 16 + lh][kk * 32 + 16 + hi * 8]);
      }
    }
    v8f sc[2];
    #pragma unroll
    for (int nt = 0; nt < 2; ++nt)
      #pragma unroll
      for (int e = 0; e < 8; ++e) sc[nt][e] = 0.0f;
    #pragma unroll
    for (int nt = 0; nt < 2; ++nt)
      #pragma unroll
      for (int kk = 0; kk < 4; ++kk)
        sc[nt] = __builtin_amdgcn_wmma_f32_16x16x32_f16(
            false, qf[kk].v, false, kb[nt * 4 + kk].v, (short)0, sc[nt], false, false);

    // Online softmax per row (row = e + 8*hi within wave's 16 rows).
    #pragma unroll
    for (int e = 0; e < 8; ++e) {
      const int qrow = qt * BR + wid * 16 + e + 8 * hi;
      const int kc0 = k0 + lh, kc1 = k0 + 16 + lh;
      const bool msk0 = kc0 > qrow, msk1 = kc1 > qrow;
      float s0 = sc[0][e] * scale;
      float s1 = sc[1][e] * scale;
      if (msk0) s0 = -3.0e38f;
      if (msk1) s1 = -3.0e38f;
      float mx = fmaxf(s0, s1);
      mx = fmaxf(mx, __shfl_xor(mx, 1, 16));
      mx = fmaxf(mx, __shfl_xor(mx, 2, 16));
      mx = fmaxf(mx, __shfl_xor(mx, 4, 16));
      mx = fmaxf(mx, __shfl_xor(mx, 8, 16));
      const float mnew = fmaxf(m_i[e], mx);
      const float mref = (mnew < -1.0e37f) ? 0.0f : mnew;
      const float corr = __expf(m_i[e] - mref);
      const float p0 = msk0 ? 0.0f : __expf(s0 - mref);
      const float p1 = msk1 ? 0.0f : __expf(s1 - mref);
      float rs = p0 + p1;
      rs += __shfl_xor(rs, 1, 16);
      rs += __shfl_xor(rs, 2, 16);
      rs += __shfl_xor(rs, 4, 16);
      rs += __shfl_xor(rs, 8, 16);
      l_i[e] = l_i[e] * corr + rs;
      m_i[e] = mnew;
      #pragma unroll
      for (int nd = 0; nd < 8; ++nd) o_acc[nd][e] *= corr;
      Ps[wid][e + 8 * hi][lh]      = (_Float16)p0;
      Ps[wid][e + 8 * hi][16 + lh] = (_Float16)p1;
    }
    __syncthreads();

    // O += P x V: preload P fragment + 8 V fragments, then 8 WMMAs.
    Frag pf;
    pf.h[0] = *(const v8h*)(&Ps[wid][lh][hi * 8]);
    pf.h[1] = *(const v8h*)(&Ps[wid][lh][hi * 8 + 16]);
    Frag vb[8];
    #pragma unroll
    for (int nd = 0; nd < 8; ++nd) {
      vb[nd].h[0] = *(const v8h*)(&Vt[nd * 16 + lh][hi * 8]);
      vb[nd].h[1] = *(const v8h*)(&Vt[nd * 16 + lh][hi * 8 + 16]);
    }
    #pragma unroll
    for (int nd = 0; nd < 8; ++nd)
      o_acc[nd] = __builtin_amdgcn_wmma_f32_16x16x32_f16(
          false, pf.v, false, vb[nd].v, (short)0, o_acc[nd], false, false);
  }

  // Normalize and write out as f16 at [b, s, head*128 + d].
  #pragma unroll
  for (int nd = 0; nd < 8; ++nd) {
    const int d = nd * 16 + lh;
    #pragma unroll
    for (int e = 0; e < 8; ++e) {
      const int qrow = qt * BR + wid * 16 + e + 8 * hi;
      const float li = l_i[e];
      const float ov = (li > 0.0f) ? o_acc[nd][e] / li : 0.0f;
      O[base + (size_t)qrow * Hdim + d] = (_Float16)ov;
    }
  }
}

// ---------------------------------------------------------------------------
// Host-side orchestration.
// ---------------------------------------------------------------------------
extern "C" void kernel_launch(void* const* d_in, const int* in_sizes, int n_in,
                              void* d_out, int out_size, void* d_ws, size_t ws_size,
                              hipStream_t stream) {
  (void)n_in; (void)out_size; (void)ws_size;

  const float* x    = (const float*)d_in[0];
  const float* g1   = (const float*)d_in[1];
  const float* be1  = (const float*)d_in[2];
  const float* cs   = (const float*)d_in[3];
  const float* sn   = (const float*)d_in[4];
  const float* w_q  = (const float*)d_in[5];
  const float* b_q  = (const float*)d_in[6];
  const float* laq  = (const float*)d_in[7];
  const float* lbq  = (const float*)d_in[8];
  const float* w_k  = (const float*)d_in[9];
  const float* b_k  = (const float*)d_in[10];
  const float* lak  = (const float*)d_in[11];
  const float* lbk  = (const float*)d_in[12];
  const float* w_v  = (const float*)d_in[13];
  const float* b_v  = (const float*)d_in[14];
  const float* lav  = (const float*)d_in[15];
  const float* lbv  = (const float*)d_in[16];
  const float* w_o  = (const float*)d_in[17];
  const float* b_o  = (const float*)d_in[18];
  const float* lao  = (const float*)d_in[19];
  const float* lbo  = (const float*)d_in[20];
  const float* g2   = (const float*)d_in[21];
  const float* be2  = (const float*)d_in[22];
  const float* w_up = (const float*)d_in[23];
  const float* b_up = (const float*)d_in[24];
  const float* laup = (const float*)d_in[25];
  const float* lbup = (const float*)d_in[26];
  const float* w_dn = (const float*)d_in[27];
  const float* b_dn = (const float*)d_in[28];
  const float* ladn = (const float*)d_in[29];
  const float* lbdn = (const float*)d_in[30];

  const int H    = in_sizes[1];          // 2048
  const int Mtot = in_sizes[0] / H;      // B*S = 4096
  const int HD   = 128;
  const int S    = in_sizes[3] / HD;     // 2048
  const int B    = Mtot / S;             // 2
  const int NH   = H / HD;               // 16
  const int FF   = in_sizes[24];         // 8192
  const int R    = in_sizes[7] / H;      // 16

  char* ws = (char*)d_ws;
  size_t off = 0;
  auto alloc = [&](size_t bytes) -> void* {
    void* p = ws + off;
    off += (bytes + 255) & ~(size_t)255;
    return p;
  };
  // f16 LoRA-folded weights
  _Float16* wq16 = (_Float16*)alloc((size_t)H * H * 2);
  _Float16* wk16 = (_Float16*)alloc((size_t)H * H * 2);
  _Float16* wv16 = (_Float16*)alloc((size_t)H * H * 2);
  _Float16* wo16 = (_Float16*)alloc((size_t)H * H * 2);
  _Float16* wup16 = (_Float16*)alloc((size_t)FF * H * 2);
  _Float16* wdn16 = (_Float16*)alloc((size_t)H * FF * 2);
  // f16 activation buffers (reused across phases)
  _Float16* bufA = (_Float16*)alloc((size_t)Mtot * H * 2); // h1, then q(rope)
  _Float16* bufB = (_Float16*)alloc((size_t)Mtot * H * 2); // q, then k(rope)
  _Float16* bufC = (_Float16*)alloc((size_t)Mtot * H * 2); // k, then attn-out
  _Float16* bufD = (_Float16*)alloc((size_t)Mtot * H * 2); // v, then h2
  _Float16* up16 = (_Float16*)alloc((size_t)Mtot * FF * 2);
  float*    x1   = (float*)alloc((size_t)Mtot * H * 4);

  const dim3 blk256(256), blk128(128);
  auto gblocks = [](size_t total) { return dim3((unsigned)((total + 255) / 256)); };

  // 1) Fold LoRA into f16 weights.
  prep_weight_kernel<<<gblocks((size_t)H * H),  blk256, 0, stream>>>(w_q,  laq,  lbq,  wq16,  H,  H,  R);
  prep_weight_kernel<<<gblocks((size_t)H * H),  blk256, 0, stream>>>(w_k,  lak,  lbk,  wk16,  H,  H,  R);
  prep_weight_kernel<<<gblocks((size_t)H * H),  blk256, 0, stream>>>(w_v,  lav,  lbv,  wv16,  H,  H,  R);
  prep_weight_kernel<<<gblocks((size_t)H * H),  blk256, 0, stream>>>(w_o,  lao,  lbo,  wo16,  H,  H,  R);
  prep_weight_kernel<<<gblocks((size_t)FF * H), blk256, 0, stream>>>(w_up, laup, lbup, wup16, FF, H,  R);
  prep_weight_kernel<<<gblocks((size_t)H * FF), blk256, 0, stream>>>(w_dn, ladn, lbdn, wdn16, H,  FF, R);

  // 2) LN1 -> h1 (f16)
  layernorm_kernel<<<dim3(Mtot), blk256, 0, stream>>>(x, g1, be1, bufA, H);

  // 3) QKV projections (WMMA GEMM, f16 out)
  const dim3 gH(H / 128, Mtot / 128);
  gemm_wmma<0><<<gH, blk256, 0, stream>>>(bufA, wq16, b_q, nullptr, bufB, Mtot, H, H);
  gemm_wmma<0><<<gH, blk256, 0, stream>>>(bufA, wk16, b_k, nullptr, bufC, Mtot, H, H);
  gemm_wmma<0><<<gH, blk256, 0, stream>>>(bufA, wv16, b_v, nullptr, bufD, Mtot, H, H);

  // 4) RoPE: q -> bufA, k -> bufB
  const size_t tot = (size_t)Mtot * H;
  rope_kernel<<<gblocks(tot), blk256, 0, stream>>>(bufB, bufA, cs, sn, S, H, tot);
  rope_kernel<<<gblocks(tot), blk256, 0, stream>>>(bufC, bufB, cs, sn, S, H, tot);

  // 5) Flash attention -> bufC (f16, [B,S,H] with heads concatenated)
  flash_attn_kernel<<<dim3(S / 64, NH, B), blk128, 0, stream>>>(bufA, bufB, bufD, bufC, S, H);

  // 6) O projection + residual -> x1 (fp32)
  gemm_wmma<2><<<gH, blk256, 0, stream>>>(bufC, wo16, b_o, x, x1, Mtot, H, H);

  // 7) LN2 -> h2 (f16)
  layernorm_kernel<<<dim3(Mtot), blk256, 0, stream>>>(x1, g2, be2, bufD, H);

  // 8) Up projection + SiLU -> up16 (f16)
  gemm_wmma<1><<<dim3(FF / 128, Mtot / 128), blk256, 0, stream>>>(
      bufD, wup16, b_up, nullptr, up16, Mtot, FF, H);

  // 9) Down projection + residual -> d_out (fp32)
  gemm_wmma<2><<<gH, blk256, 0, stream>>>(up16, wdn16, b_dn, x1, (float*)d_out, Mtot, H, FF);
}
